// LPPMLP_58162447122879
// MI455X (gfx1250) — matrix operand order repaired and checked by reading, not compiled, not run on hardware
//
#include <hip/hip_runtime.h>
#include <hip/hip_bf16.h>

// ---------------------------------------------------------------------------
// Problem constants (fixed by the reference implementation)
// ---------------------------------------------------------------------------
#define IN_F     64
#define MID_F    128
#define OUT_F    32
#define NROWS    1048576
#define NUM_SEG  100000

#define ROWS_PER_BLOCK 128          // 8 waves x 16 rows
#define WAVES_PER_BLOCK 8
#define BLOCKS (NROWS / ROWS_PER_BLOCK)

typedef __attribute__((ext_vector_type(16))) __bf16       v16bf;
typedef __attribute__((ext_vector_type(8)))  float        v8f;
typedef __attribute__((ext_vector_type(4)))  unsigned int v4u;
typedef __attribute__((ext_vector_type(4)))  float        v4f;

union AFrag {
    v16bf  v;
    __bf16 h[16];
    v4u    q[2];
};

// native f32 -> bf16 (round-to-nearest-even via hardware cvt)
__device__ __forceinline__ __bf16 f2bf(float f) { return (__bf16)f; }

// ---------------------------------------------------------------------------
// Kernel 0: zero the sum buffer (d_out) and the count buffer (d_ws)
// ---------------------------------------------------------------------------
__global__ void zero_kernel(float* __restrict__ out, float* __restrict__ counts) {
    int i = blockIdx.x * blockDim.x + threadIdx.x;
    if (i < NUM_SEG * OUT_F) out[i]    = 0.0f;
    if (i < NUM_SEG)         counts[i] = 0.0f;
}

// ---------------------------------------------------------------------------
// Kernel 1: fused bf16-WMMA MLP + scatter-add
// ---------------------------------------------------------------------------
__global__ __launch_bounds__(256, 2)
void mlp_scatter_kernel(const float* __restrict__ vecs,
                        const int*   __restrict__ sidx,
                        const float* __restrict__ W1,
                        const float* __restrict__ b1,
                        const float* __restrict__ W2,
                        const float* __restrict__ b2,
                        float* __restrict__ out,     // [NUM_SEG, 32] running sums
                        float* __restrict__ counts)  // [NUM_SEG]
{
    // --- LDS: weights in WMMA B-fragment layout (bf16), biases, h buffer ----
    // W1 fragments: 8 ntiles x 2 ktiles x 32 lanes x 16 elems = 8192 bf16
    __shared__ __align__(32) __bf16 w1f[16 * 32 * 16];
    // W2 fragments: 4 ktiles x 2 ntiles x 32 lanes x 16 elems = 4096 bf16
    __shared__ __align__(32) __bf16 w2f[8 * 32 * 16];
    // per-wave 16x128 bf16 activation tile
    __shared__ __align__(32) __bf16 hbuf[WAVES_PER_BLOCK * 16 * MID_F];
    __shared__ float b1s[MID_F];
    __shared__ float b2s[OUT_F];

    const int tid  = threadIdx.x;
    const int lane = tid & 31;
    const int wave = tid >> 5;

    // ---- repack W1 (64x128 f32 row-major) into B-fragment layout ----------
    // B tile layout (32x16, bf16): lane<16 holds col=lane, K=kbase+0..7,16..23
    //                              lane>=16 holds col=lane-16, K=kbase+8..15,24..31
    #pragma unroll
    for (int rep = 0; rep < 2; ++rep) {
        int p     = tid + rep * 256;          // 512 (tile,lane) pairs total
        int tile  = p >> 5;                   // 0..15  (= ntile*2 + ktile)
        int l     = p & 31;
        int ntile = tile >> 1;
        int ktile = tile & 1;
        int kb    = ktile * 32 + (l >> 4) * 8;
        int n     = ntile * 16 + (l & 15);
        #pragma unroll
        for (int e = 0; e < 16; ++e)
            w1f[(tile * 32 + l) * 16 + e] = f2bf(W1[(kb + (e & 7) + ((e >> 3) << 4)) * MID_F + n]);
    }

    // ---- repack W2 (128x32 f32 row-major): 256 (tile,lane) pairs ----------
    {
        int tile  = tid >> 5;                 // 0..7  (= ktile*2 + ntile)
        int l     = tid & 31;
        int ktile = tile >> 1;
        int ntile = tile & 1;
        int kb    = ktile * 32 + (l >> 4) * 8;
        int n     = ntile * 16 + (l & 15);
        #pragma unroll
        for (int e = 0; e < 16; ++e)
            w2f[(tile * 32 + l) * 16 + e] = f2bf(W2[(kb + (e & 7) + ((e >> 3) << 4)) * OUT_F + n]);
    }

    if (tid < MID_F) b1s[tid] = b1[tid];
    if (tid < OUT_F) b2s[tid] = b2[tid];

    const int rowBase = blockIdx.x * ROWS_PER_BLOCK + wave * 16;
    const int myRow   = rowBase + (lane & 15);

    // prefetch this wave's input rows while the repack settles
    __builtin_prefetch(vecs + (size_t)myRow * IN_F, 0, 0);

    __syncthreads();

    // ---- layer 1: A fragments (16x32 bf16) straight from global fp32 ------
    // A layout: lane<16 -> elems 0..7 = K kt*32+0..7,  elems 8..15 = K kt*32+16..23
    //           lane>=16 -> elems 0..7 = K kt*32+8..15, elems 8..15 = K kt*32+24..31
    AFrag a[2];
    const v4f* vrow = reinterpret_cast<const v4f*>(vecs + (size_t)myRow * IN_F);
    #pragma unroll
    for (int kt = 0; kt < 2; ++kt) {
        int q0 = (kt * 32 + (lane >> 4) * 8) >> 2;   // float4 index of first run
        v4f x0 = vrow[q0];
        v4f x1 = vrow[q0 + 1];
        v4f x2 = vrow[q0 + 4];                        // +16 floats
        v4f x3 = vrow[q0 + 5];
        #pragma unroll
        for (int e = 0; e < 4; ++e) {
            a[kt].h[e]      = f2bf(x0[e]);
            a[kt].h[4 + e]  = f2bf(x1[e]);
            a[kt].h[8 + e]  = f2bf(x2[e]);
            a[kt].h[12 + e] = f2bf(x3[e]);
        }
    }

    const int nCol = lane & 15;               // accumulator column within tile
    const int m8   = 8 * (lane >> 4);         // accumulator row-half offset
    __bf16* hrowBase = &hbuf[wave * 16 * MID_F];

    #pragma unroll
    for (int nt = 0; nt < 8; ++nt) {
        v8f c = {};
        #pragma unroll
        for (int kt = 0; kt < 2; ++kt) {
            v16bf b = *reinterpret_cast<const v16bf*>(
                &w1f[((nt * 2 + kt) * 32 + lane) * 16]);
            c = __builtin_amdgcn_wmma_f32_16x16x32_bf16(
                    false, a[kt].v, false, b, (short)0, c, false, false);
        }
        // bias + ReLU + bf16 store to per-wave h tile
        float bv = b1s[nt * 16 + nCol];
        #pragma unroll
        for (int r = 0; r < 8; ++r) {
            float x = c[r] + bv;
            x = x > 0.0f ? x : 0.0f;
            hrowBase[(r + m8) * MID_F + nt * 16 + nCol] = f2bf(x);
        }
    }

    // ---- layer 2: h (16x128 bf16 in LDS) x W2 -> 16x32 --------------------
    v8f c2[2] = {{}, {}};
    #pragma unroll
    for (int kt = 0; kt < 4; ++kt) {
        AFrag af;
        int kb0 = kt * 32 + (lane >> 4) * 8;
        const __bf16* hr = &hrowBase[(lane & 15) * MID_F];
        af.q[0] = *reinterpret_cast<const v4u*>(&hr[kb0]);
        af.q[1] = *reinterpret_cast<const v4u*>(&hr[kb0 + 16]);
        #pragma unroll
        for (int nt = 0; nt < 2; ++nt) {
            v16bf b = *reinterpret_cast<const v16bf*>(
                &w2f[((kt * 2 + nt) * 32 + lane) * 16]);
            c2[nt] = __builtin_amdgcn_wmma_f32_16x16x32_bf16(
                         false, af.v, false, b, (short)0, c2[nt], false, false);
        }
    }

    // ---- scatter-add preds into segment sums ------------------------------
    int seg[8];
    #pragma unroll
    for (int r = 0; r < 8; ++r)
        seg[r] = sidx[rowBase + m8 + r];

    #pragma unroll
    for (int nt = 0; nt < 2; ++nt) {
        float bv = b2s[nt * 16 + nCol];
        #pragma unroll
        for (int r = 0; r < 8; ++r) {
            float p = c2[nt][r] + bv;
            atomicAdd(&out[(size_t)seg[r] * OUT_F + nt * 16 + nCol], p);
        }
    }
    // one count per row (lanes 0..15 cover the 16 rows of this wave)
    if (lane < 16)
        atomicAdd(&counts[sidx[rowBase + lane]], 1.0f);
}

// ---------------------------------------------------------------------------
// Kernel 2: divide sums by max(count, 1)
// ---------------------------------------------------------------------------
__global__ void finalize_kernel(float* __restrict__ out,
                                const float* __restrict__ counts) {
    int i = blockIdx.x * blockDim.x + threadIdx.x;
    if (i < NUM_SEG * OUT_F) {
        float c = counts[i >> 5];            // OUT_F == 32
        out[i] = out[i] / fmaxf(c, 1.0f);
    }
}

// ---------------------------------------------------------------------------
extern "C" void kernel_launch(void* const* d_in, const int* in_sizes, int n_in,
                              void* d_out, int out_size, void* d_ws, size_t ws_size,
                              hipStream_t stream) {
    (void)in_sizes; (void)n_in; (void)out_size; (void)ws_size;
    const float* vecs = (const float*)d_in[0];
    const int*   sidx = (const int*)d_in[1];
    // d_in[2] = num_segments scalar (compile-time constant here)
    const float* W1   = (const float*)d_in[3];
    const float* b1   = (const float*)d_in[4];
    const float* W2   = (const float*)d_in[5];
    const float* b2   = (const float*)d_in[6];

    float* out    = (float*)d_out;
    float* counts = (float*)d_ws;            // NUM_SEG floats of scratch

    {
        int total = NUM_SEG * OUT_F;         // covers counts too (NUM_SEG < total)
        zero_kernel<<<(total + 255) / 256, 256, 0, stream>>>(out, counts);
    }
    mlp_scatter_kernel<<<BLOCKS, 256, 0, stream>>>(vecs, sidx, W1, b1, W2, b2,
                                                   out, counts);
    {
        int total = NUM_SEG * OUT_F;
        finalize_kernel<<<(total + 255) / 256, 256, 0, stream>>>(out, counts);
    }
}
